// Encoder_84971632984653
// MI455X (gfx1250) — matrix operand order repaired and checked by reading, not compiled, not run on hardware
//
#include <hip/hip_runtime.h>
#include <hip/hip_bf16.h>
#include <stdint.h>

// ---------------------------------------------------------------------------
// Types
// ---------------------------------------------------------------------------
typedef __bf16 bf16;
typedef __attribute__((ext_vector_type(16))) __bf16 v16bf;
typedef __attribute__((ext_vector_type(8)))  __bf16 v8bf;
typedef __attribute__((ext_vector_type(8)))  float  v8f;
typedef __attribute__((ext_vector_type(4)))  int    v4i;

// Dimensions (fixed by the reference)
#define BATCH 8
#define SEQ   1024
#define DMODEL 1024
#define NHEAD 16
#define HDIM  64
#define FFDIM 2048
#define NTOK  (BATCH * SEQ)   // 8192

// ---------------------------------------------------------------------------
// CDNA5 async global->LDS copy (guarded: falls back to reg staging if the
// toolchain doesn't expose the gfx1250 async builtins).
// Probe-verified signature: (v4i* gaddr, v4i* ldsaddr, imm offset, imm cpol).
// ---------------------------------------------------------------------------
#if __has_builtin(__builtin_amdgcn_global_load_async_to_lds_b128) && \
    __has_builtin(__builtin_amdgcn_s_wait_asynccnt)
#define HAS_ASYNC 1
#else
#define HAS_ASYNC 0
#endif

#if HAS_ASYNC
__device__ __forceinline__ void async_cp16(const bf16* g, bf16* l) {
    __builtin_amdgcn_global_load_async_to_lds_b128((v4i*)g, (v4i*)l, 0, 0);
}
#endif

// ---------------------------------------------------------------------------
// WMMA fragment loaders (CDNA5 16-bit layouts, wave32)
// ---------------------------------------------------------------------------
__device__ __forceinline__ v16bf load_a_frag(const bf16* base, int ld, int lane) {
    int m  = lane & 15;
    int kb = (lane >> 4) << 3;
    const bf16* p = base + (size_t)m * ld + kb;
    union { v16bf v; v8bf h[2]; } u;
    u.h[0] = *(const v8bf*)(p);
    u.h[1] = *(const v8bf*)(p + 16);
    return u.v;
}

// src is B^T stored row-major [N][K] with row stride ld (elements)
__device__ __forceinline__ v16bf load_bt_frag(const bf16* base, int ld, int lane) {
    int n  = lane & 15;
    int kb = (lane >> 4) << 3;
    const bf16* p = base + (size_t)n * ld + kb;
    union { v16bf v; v8bf h[2]; } u;
    u.h[0] = *(const v8bf*)(p);
    u.h[1] = *(const v8bf*)(p + 16);
    return u.v;
}

// ---------------------------------------------------------------------------
// Elementwise f32 -> bf16 cast (activations)
// ---------------------------------------------------------------------------
__global__ __launch_bounds__(256) void cast_bf16_kernel(const float* __restrict__ in,
                                                        bf16* __restrict__ out, int n) {
    int i = (blockIdx.x * 256 + threadIdx.x) * 4;
    if (i + 3 < n) {
        float4 v = *(const float4*)(in + i);
        out[i + 0] = (bf16)v.x; out[i + 1] = (bf16)v.y;
        out[i + 2] = (bf16)v.z; out[i + 3] = (bf16)v.w;
    } else {
        for (int k = i; k < n; ++k) out[k] = (bf16)in[k];
    }
}

// ---------------------------------------------------------------------------
// Transpose + cast: W[K][N] f32 -> Wt[N][K] bf16 (one-time, so GEMM B tiles
// stage with vector b128 ops instead of 16 scalar ds_store_b16 per thread)
// ---------------------------------------------------------------------------
__global__ __launch_bounds__(256) void transpose_cast_kernel(
    const float* __restrict__ W, bf16* __restrict__ Wt, int K, int N)
{
    __shared__ float t[32][33];
    const int tx = threadIdx.x & 31;
    const int ty = threadIdx.x >> 5;          // 0..7
    const int n0 = blockIdx.x * 32;
    const int k0 = blockIdx.y * 32;
#pragma unroll
    for (int j = 0; j < 4; ++j)
        t[ty + 8 * j][tx] = W[(size_t)(k0 + ty + 8 * j) * N + n0 + tx];
    __syncthreads();
#pragma unroll
    for (int j = 0; j < 4; ++j)
        Wt[(size_t)(n0 + ty + 8 * j) * K + k0 + tx] = (bf16)t[tx][ty + 8 * j];
}

// ---------------------------------------------------------------------------
// Tiled WMMA GEMM:  C[M,N] = A[M,K](bf16) * Bt[N,K](bf16)^T + bias, opt ReLU.
// Block tile 128x128, BK=32, 256 threads = 8 waves, wave tile 32x64.
// Double-buffered LDS; async global->LDS staging when available.
// ---------------------------------------------------------------------------
#define BM 128
#define BN 128
#define BK 32
#define LDT 40   // padded LDS K-stride (80B rows: 16B aligned, conflict-friendly)

__global__ __launch_bounds__(256) void gemm_bf16_kernel(
    const bf16* __restrict__ A, const bf16* __restrict__ Bt,
    const float* __restrict__ bias, float* __restrict__ Cf, bf16* __restrict__ Cb,
    int M, int N, int K, int relu)
{
    __shared__ bf16 As[2][BM][LDT];
    __shared__ bf16 Bs[2][BN][LDT];

    const int tid  = threadIdx.x;
    const int lane = tid & 31;
    const int w    = tid >> 5;
    const int bm   = blockIdx.y * BM;
    const int bn   = blockIdx.x * BN;
    const int mo   = (w & 3) * 32;   // wave row offset in tile
    const int no   = (w >> 2) * 64;  // wave col offset in tile
    const int r    = tid >> 1;              // staging row 0..127
    const int cc   = (tid & 1) * 16;        // staging K-chunk

    v8f acc[2][4] = {};
    const int nIter = K / BK;

    auto compute = [&](int cur) {
        v16bf a0 = load_a_frag(&As[cur][mo][0],      LDT, lane);
        v16bf a1 = load_a_frag(&As[cur][mo + 16][0], LDT, lane);
#pragma unroll
        for (int t = 0; t < 4; ++t) {
            v16bf bfr = load_bt_frag(&Bs[cur][no + t * 16][0], LDT, lane);
            acc[0][t] = __builtin_amdgcn_wmma_f32_16x16x32_bf16(
                false, a0, false, bfr, (short)0, acc[0][t], false, false);
            acc[1][t] = __builtin_amdgcn_wmma_f32_16x16x32_bf16(
                false, a1, false, bfr, (short)0, acc[1][t], false, false);
        }
    };

#if HAS_ASYNC
    auto stage = [&](int k0, int buf) {
        const bf16* ga = A  + (size_t)(bm + r) * K + k0 + cc;
        const bf16* gb = Bt + (size_t)(bn + r) * K + k0 + cc;
        async_cp16(ga,     &As[buf][r][cc]);
        async_cp16(ga + 8, &As[buf][r][cc + 8]);
        async_cp16(gb,     &Bs[buf][r][cc]);
        async_cp16(gb + 8, &Bs[buf][r][cc + 8]);
    };
    stage(0, 0);
    for (int i = 0; i < nIter; ++i) {
        const int cur = i & 1;
        if (i + 1 < nIter) {
            stage((i + 1) * BK, cur ^ 1);
            __builtin_amdgcn_s_wait_asynccnt(4);   // current tile's 4 copies done
        } else {
            __builtin_amdgcn_s_wait_asynccnt(0);
        }
        __syncthreads();          // whole tile complete across all waves
        compute(cur);
        __syncthreads();          // tile buffer may be overwritten next iter
    }
#else
    // Register-staged double buffer fallback
    v8bf ra0, ra1, rb0, rb1;
    auto gload = [&](int k0) {
        const bf16* ga = A  + (size_t)(bm + r) * K + k0 + cc;
        const bf16* gb = Bt + (size_t)(bn + r) * K + k0 + cc;
        ra0 = *(const v8bf*)(ga); ra1 = *(const v8bf*)(ga + 8);
        rb0 = *(const v8bf*)(gb); rb1 = *(const v8bf*)(gb + 8);
    };
    auto sstore = [&](int buf) {
        *(v8bf*)&As[buf][r][cc] = ra0; *(v8bf*)&As[buf][r][cc + 8] = ra1;
        *(v8bf*)&Bs[buf][r][cc] = rb0; *(v8bf*)&Bs[buf][r][cc + 8] = rb1;
    };
    gload(0);
    sstore(0);
    for (int i = 0; i < nIter; ++i) {
        const int cur = i & 1;
        __syncthreads();                      // tile cur visible
        if (i + 1 < nIter) gload((i + 1) * BK);
        compute(cur);
        __syncthreads();                      // all waves done reading nxt buffer
        if (i + 1 < nIter) sstore(cur ^ 1);
    }
#endif

    // ---- epilogue: bias (+ReLU) and dual-precision store ----
    const int nn = lane & 15;
    const int mb = (lane >> 4) * 8;
#pragma unroll
    for (int i = 0; i < 2; ++i) {
#pragma unroll
        for (int t = 0; t < 4; ++t) {
            int col = bn + no + t * 16 + nn;
            float bv = bias ? bias[col] : 0.0f;
#pragma unroll
            for (int j = 0; j < 8; ++j) {
                int row = bm + mo + i * 16 + mb + j;
                float val = acc[i][t][j] + bv;
                if (relu) val = fmaxf(val, 0.0f);
                if (Cf) Cf[(size_t)row * N + col] = val;
                if (Cb) Cb[(size_t)row * N + col] = (bf16)val;
            }
        }
    }
}

// ---------------------------------------------------------------------------
// Flash attention: one block per (b, h, 128 query rows); 8 waves x 16 q-rows.
// ---------------------------------------------------------------------------
__global__ __launch_bounds__(256) void attn_kernel(
    const bf16* __restrict__ Qg, const bf16* __restrict__ Kg,
    const bf16* __restrict__ Vg, bf16* __restrict__ ctxb)
{
    __shared__ bf16 Kl[32][72];      // [key][dim]   (144B rows, 16B aligned)
    __shared__ bf16 Vt[64][40];      // [vcol][key]  (80B rows)
    __shared__ bf16 Pl[8][16][40];   // per-wave P staging [qrow][key]

    const int tid  = threadIdx.x;
    const int lane = tid & 31;
    const int w    = tid >> 5;

    const int qb = blockIdx.x & 7;
    const int bh = blockIdx.x >> 3;
    const int b  = bh >> 4;
    const int h  = bh & 15;
    const int rowbase = b * SEQ;
    const int colbase = h * HDIM;
    const int q0 = qb * 128 + w * 16;

    // ---- load Q fragments (16 rows x 64 dims -> two 16x32 A-frags) ----
    const int am  = lane & 15;
    const int akb = (lane >> 4) * 8;
    union { v16bf v; v8bf h8[2]; } qf[2];
#pragma unroll
    for (int c = 0; c < 2; ++c) {
        const bf16* p = Qg + (size_t)(rowbase + q0 + am) * DMODEL + colbase + c * 32 + akb;
        qf[c].h8[0] = *(const v8bf*)(p);
        qf[c].h8[1] = *(const v8bf*)(p + 16);
    }

    float mrun[8], lrun[8];
#pragma unroll
    for (int j = 0; j < 8; ++j) { mrun[j] = -1e30f; lrun[j] = 0.0f; }
    v8f o[4] = {};

    const int nn = lane & 15;
    const int mb = (lane >> 4) * 8;

    for (int kb = 0; kb < SEQ; kb += 32) {
        __syncthreads();
        // ---- cooperative stage of K tile (async when available) + V^T ----
        {
            int key = tid >> 3, ch = tid & 7;
            const bf16* kp = Kg + (size_t)(rowbase + kb + key) * DMODEL + colbase + ch * 8;
#if HAS_ASYNC
            async_cp16(kp, &Kl[key][ch * 8]);
#else
            *(v8bf*)&Kl[key][ch * 8] = *(const v8bf*)kp;
#endif
            const bf16* vp = Vg + (size_t)(rowbase + kb + key) * DMODEL + colbase + ch * 8;
            v8bf vv = *(const v8bf*)vp;
#pragma unroll
            for (int e = 0; e < 8; ++e) Vt[ch * 8 + e][key] = vv[e];
        }
#if HAS_ASYNC
        __builtin_amdgcn_s_wait_asynccnt(0);
#endif
        __syncthreads();

        // ---- scores: S[16 x 32] = Q(16x64) . K^T, two 16-key halves ----
        v8f sc[2];
#pragma unroll
        for (int half = 0; half < 2; ++half) {
            v8f s = {};
#pragma unroll
            for (int c = 0; c < 2; ++c) {
                union { v16bf v; v8bf h8[2]; } u;
                const bf16* p = &Kl[half * 16 + nn][c * 32 + ((lane >> 4) * 8)];
                u.h8[0] = *(const v8bf*)(p);
                u.h8[1] = *(const v8bf*)(p + 16);
                s = __builtin_amdgcn_wmma_f32_16x16x32_bf16(
                    false, qf[c].v, false, u.v, (short)0, s, false, false);
            }
            sc[half] = s;
        }

        // ---- online softmax (f32), rows mb..mb+7 per lane half ----
        float p0[8], p1[8];
#pragma unroll
        for (int j = 0; j < 8; ++j) {
            float a  = sc[0][j] * 0.125f;   // 1/sqrt(64)
            float bb = sc[1][j] * 0.125f;
            float rm = fmaxf(a, bb);
#pragma unroll
            for (int msk = 1; msk < 16; msk <<= 1)
                rm = fmaxf(rm, __shfl_xor(rm, msk, 32));
            float mn    = fmaxf(mrun[j], rm);
            float alpha = __expf(mrun[j] - mn);
            p0[j] = __expf(a - mn);
            p1[j] = __expf(bb - mn);
            float rs = p0[j] + p1[j];
#pragma unroll
            for (int msk = 1; msk < 16; msk <<= 1)
                rs += __shfl_xor(rs, msk, 32);
            lrun[j] = lrun[j] * alpha + rs;
            mrun[j] = mn;
#pragma unroll
            for (int t = 0; t < 4; ++t) o[t][j] *= alpha;
        }

        // ---- stage P (C-layout -> LDS) then load as 16x32 A-frag ----
#pragma unroll
        for (int j = 0; j < 8; ++j) {
            Pl[w][mb + j][nn]      = (bf16)p0[j];
            Pl[w][mb + j][16 + nn] = (bf16)p1[j];
        }
        v16bf pA = load_a_frag(&Pl[w][0][0], 40, lane);

        // ---- O += P(16x32) . V(32x64) ----
#pragma unroll
        for (int t = 0; t < 4; ++t) {
            union { v16bf v; v8bf h8[2]; } u;
            const bf16* p = &Vt[t * 16 + nn][(lane >> 4) * 8];
            u.h8[0] = *(const v8bf*)(p);
            u.h8[1] = *(const v8bf*)(p + 16);
            o[t] = __builtin_amdgcn_wmma_f32_16x16x32_bf16(
                false, pA, false, u.v, (short)0, o[t], false, false);
        }
    }

    // ---- normalize and write ctx (bf16, concat-head layout [tok, H*64]) ----
#pragma unroll
    for (int j = 0; j < 8; ++j) {
        float inv = 1.0f / lrun[j];
        int row = rowbase + q0 + mb + j;
#pragma unroll
        for (int t = 0; t < 4; ++t) {
            int col = colbase + t * 16 + nn;
            ctxb[(size_t)row * DMODEL + col] = (bf16)(o[t][j] * inv);
        }
    }
}

// ---------------------------------------------------------------------------
// LayerNorm over last dim (1024): out = LN(a + b) * g + beta.
// ---------------------------------------------------------------------------
__global__ __launch_bounds__(256) void ln_kernel(
    const float* __restrict__ A, const float* __restrict__ Bv,
    const float* __restrict__ g, const float* __restrict__ be,
    float* __restrict__ outF, bf16* __restrict__ outB)
{
    __shared__ float rs[256], rq[256];
    const int row = blockIdx.x;
    const int tid = threadIdx.x;
    const float* a  = A  + (size_t)row * DMODEL;
    const float* bb = Bv + (size_t)row * DMODEL;

    float v[4], s = 0.0f, sq = 0.0f;
#pragma unroll
    for (int i = 0; i < 4; ++i) {
        int c = tid + i * 256;
        v[i] = a[c] + bb[c];
        s  += v[i];
        sq += v[i] * v[i];
    }
    rs[tid] = s; rq[tid] = sq;
    __syncthreads();
    for (int st = 128; st > 0; st >>= 1) {
        if (tid < st) { rs[tid] += rs[tid + st]; rq[tid] += rq[tid + st]; }
        __syncthreads();
    }
    float mean = rs[0] * (1.0f / DMODEL);
    float var  = rq[0] * (1.0f / DMODEL) - mean * mean;
    float rstd = rsqrtf(var + 1e-5f);
#pragma unroll
    for (int i = 0; i < 4; ++i) {
        int c = tid + i * 256;
        float ov = (v[i] - mean) * rstd * g[c] + be[c];
        outF[(size_t)row * DMODEL + c] = ov;
        if (outB) outB[(size_t)row * DMODEL + c] = (bf16)ov;
    }
}

// ---------------------------------------------------------------------------
// Host-side orchestration
// ---------------------------------------------------------------------------
extern "C" void kernel_launch(void* const* d_in, const int* in_sizes, int n_in,
                              void* d_out, int out_size, void* d_ws, size_t ws_size,
                              hipStream_t stream) {
    (void)in_sizes; (void)n_in; (void)out_size; (void)ws_size;

    const float* x     = (const float*)d_in[0];
    const float* Wq    = (const float*)d_in[1];
    const float* bq    = (const float*)d_in[2];
    const float* Wk    = (const float*)d_in[3];
    const float* bk    = (const float*)d_in[4];
    const float* Wv    = (const float*)d_in[5];
    const float* bv    = (const float*)d_in[6];
    const float* Wo    = (const float*)d_in[7];
    const float* bo    = (const float*)d_in[8];
    const float* ln1g  = (const float*)d_in[9];
    const float* ln1b  = (const float*)d_in[10];
    const float* W1    = (const float*)d_in[11];
    const float* b1    = (const float*)d_in[12];
    const float* W2    = (const float*)d_in[13];
    const float* b2    = (const float*)d_in[14];
    const float* ln2g  = (const float*)d_in[15];
    const float* ln2b  = (const float*)d_in[16];
    float* out = (float*)d_out;

    char* ws = (char*)d_ws;
    const size_t MB = 1ull << 20;
    bf16*  xb   = (bf16*)(ws + 0   * MB);  // 16 MB  [M][K]
    bf16*  WqT  = (bf16*)(ws + 16  * MB);  // 2 MB   [N][K]
    bf16*  WkT  = (bf16*)(ws + 18  * MB);
    bf16*  WvT  = (bf16*)(ws + 20  * MB);
    bf16*  WoT  = (bf16*)(ws + 22  * MB);
    bf16*  W1T  = (bf16*)(ws + 24  * MB);  // 4 MB
    bf16*  W2T  = (bf16*)(ws + 28  * MB);  // 4 MB
    bf16*  Qb   = (bf16*)(ws + 32  * MB);  // 16 MB
    bf16*  Kb   = (bf16*)(ws + 48  * MB);  // 16 MB
    bf16*  Vb   = (bf16*)(ws + 64  * MB);  // 16 MB
    bf16*  ctxb = (bf16*)(ws + 80  * MB);  // 16 MB
    float* cF   = (float*)(ws + 96  * MB); // 32 MB
    float* hF   = (float*)(ws + 128 * MB); // 32 MB
    bf16*  hB   = (bf16*)(ws + 160 * MB);  // 16 MB
    bf16*  actB = (bf16*)(ws + 32  * MB);  // 32 MB, aliases Qb/Kb (dead by then)
    float* yF   = (float*)(ws + 64  * MB); // 32 MB, aliases Vb/ctxb (dead by then)

    // 1) precision conversion; weights transposed to [N][K] once
    cast_bf16_kernel<<<(NTOK * DMODEL + 1023) / 1024, 256, 0, stream>>>(x, xb, NTOK * DMODEL);
    auto tcast = [&](const float* W, bf16* Wt, int K, int N) {
        transpose_cast_kernel<<<dim3(N / 32, K / 32), 256, 0, stream>>>(W, Wt, K, N);
    };
    tcast(Wq, WqT, DMODEL, DMODEL);
    tcast(Wk, WkT, DMODEL, DMODEL);
    tcast(Wv, WvT, DMODEL, DMODEL);
    tcast(Wo, WoT, DMODEL, DMODEL);
    tcast(W1, W1T, DMODEL, FFDIM);
    tcast(W2, W2T, FFDIM, DMODEL);

    // 2) QKV projections
    dim3 gQ(DMODEL / BN, NTOK / BM);
    gemm_bf16_kernel<<<gQ, 256, 0, stream>>>(xb, WqT, bq, nullptr, Qb,
                                             NTOK, DMODEL, DMODEL, 0);
    gemm_bf16_kernel<<<gQ, 256, 0, stream>>>(xb, WkT, bk, nullptr, Kb,
                                             NTOK, DMODEL, DMODEL, 0);
    gemm_bf16_kernel<<<gQ, 256, 0, stream>>>(xb, WvT, bv, nullptr, Vb,
                                             NTOK, DMODEL, DMODEL, 0);

    // 3) fused flash attention -> ctx (concat heads)
    attn_kernel<<<BATCH * NHEAD * (SEQ / 128), 256, 0, stream>>>(Qb, Kb, Vb, ctxb);

    // 4) output projection: c = ctx @ Wo + bo   (f32)
    gemm_bf16_kernel<<<gQ, 256, 0, stream>>>(ctxb, WoT, bo, cF, nullptr,
                                             NTOK, DMODEL, DMODEL, 0);

    // 5) h = LN(x + c)   (f32 + bf16 mirror)
    ln_kernel<<<NTOK, 256, 0, stream>>>(x, cF, ln1g, ln1b, hF, hB);

    // 6) FFN: act = relu(h @ W1 + b1)  (bf16)
    dim3 gF1(FFDIM / BN, NTOK / BM);
    gemm_bf16_kernel<<<gF1, 256, 0, stream>>>(hB, W1T, b1, nullptr, actB,
                                              NTOK, FFDIM, DMODEL, 1);

    // 7) y = act @ W2 + b2   (f32)
    gemm_bf16_kernel<<<gQ, 256, 0, stream>>>(actB, W2T, b2, yF, nullptr,
                                             NTOK, DMODEL, FFDIM, 0);

    // 8) out = LN(h + y)
    ln_kernel<<<NTOK, 256, 0, stream>>>(hF, yF, ln2g, ln2b, out, nullptr);
}
